// CLSPostProcessor_19937238188275
// MI455X (gfx1250) — compile-verified, matrix operand order checked by reading, not run on hardware
//
#include <hip/hip_runtime.h>
#include <hip/hip_bf16.h>

// ---------------------------------------------------------------------------
// Detection post-process for MI455X (gfx1250, wave32).
// Pipeline: softmax+threshold compaction -> 2-level radix top-K threshold ->
// exact select -> in-LDS bitonic sort (desc) -> gather/clip boxes ->
// fused WMMA(one-hot Gram = same-class mask) + division-free IoU tile kernel
// -> 2MB suppression bitmask -> one-workgroup bit-parallel NMS scan -> out.
// ---------------------------------------------------------------------------

#define NROWS   32768
#define NCLS    81
#define KSEL    4096
#define MCAP    622592          // hard bound: <=19 probs/row can exceed 0.05
#define IMGW    1333.0f
#define IMGH    800.0f
#define SCORE_T 0.05f
#define NMS_T   0.5f

typedef __attribute__((ext_vector_type(16))) _Float16 v16h;
typedef __attribute__((ext_vector_type(8)))  float    v8f;

// counters: [0]=candCount [1]=selCount [2]=tieCount [3]=bucketB [4]=aboveB
//           [5]=keyT [6]=tiesNeeded
// ---------------------------------------------------------------------------
__global__ __launch_bounds__(256) void init_ws(unsigned* counters, unsigned* h1,
                                               unsigned* h2, float* selScore,
                                               unsigned* selIdx) {
  int i = blockIdx.x * 256 + threadIdx.x;
  if (i < 65536) { h1[i] = 0u; h2[i] = 0u; }
  if (i < KSEL)  { selScore[i] = -1.0f; selIdx[i] = 0u; }
  if (i < 16)    counters[i] = 0u;
}

// one wave per row: softmax over 81 classes, emit candidates > 0.05 (cls != 0)
__global__ __launch_bounds__(256) void softmax_compact(const float* __restrict__ x,
                                                       unsigned* counters,
                                                       unsigned* candKey,
                                                       unsigned* candIdx) {
  int wid  = threadIdx.x >> 5;
  int lane = threadIdx.x & 31;
  int row  = blockIdx.x * 8 + wid;
  if (row >= NROWS) return;
  const float* xr = x + (size_t)row * NCLS;
  float v0 = xr[lane];
  float v1 = xr[lane + 32];
  float v2 = (lane + 64 < NCLS) ? xr[lane + 64] : -1e30f;
  float m = fmaxf(v0, fmaxf(v1, v2));
  #pragma unroll
  for (int o = 16; o > 0; o >>= 1) m = fmaxf(m, __shfl_xor(m, o, 32));
  float e0 = expf(v0 - m), e1 = expf(v1 - m);
  float e2 = (lane + 64 < NCLS) ? expf(v2 - m) : 0.0f;
  float s = e0 + e1 + e2;
  #pragma unroll
  for (int o = 16; o > 0; o >>= 1) s += __shfl_xor(s, o, 32);
  float inv = 1.0f / s;
  float p0 = e0 * inv, p1 = e1 * inv, p2 = e2 * inv;
  if (lane != 0 && p0 > SCORE_T) {
    unsigned pos = atomicAdd(&counters[0], 1u);
    if (pos < MCAP) { candKey[pos] = __float_as_uint(p0); candIdx[pos] = row * NCLS + lane; }
  }
  if (p1 > SCORE_T) {
    unsigned pos = atomicAdd(&counters[0], 1u);
    if (pos < MCAP) { candKey[pos] = __float_as_uint(p1); candIdx[pos] = row * NCLS + lane + 32; }
  }
  if (lane + 64 < NCLS && p2 > SCORE_T) {
    unsigned pos = atomicAdd(&counters[0], 1u);
    if (pos < MCAP) { candKey[pos] = __float_as_uint(p2); candIdx[pos] = row * NCLS + lane + 64; }
  }
}

__global__ __launch_bounds__(256) void hist_hi(const unsigned* __restrict__ candKey,
                                               const unsigned* counters, unsigned* h1) {
  unsigned M = counters[0]; if (M > MCAP) M = MCAP;
  unsigned i = blockIdx.x * 256 + threadIdx.x;
  if (i < M) atomicAdd(&h1[candKey[i] >> 16], 1u);
}

__global__ __launch_bounds__(1024) void scan_hi(const unsigned* __restrict__ h1,
                                                unsigned* counters) {
  __shared__ unsigned chunk[1024];
  int t = threadIdx.x;
  unsigned s = 0;
  for (int i = 0; i < 64; ++i) s += h1[t * 64 + i];
  chunk[t] = s;
  __syncthreads();
  if (t == 0) {
    unsigned cum = 0; int B = -1; unsigned above = 0;
    for (int ci = 1023; ci >= 0 && B < 0; --ci) {
      if (cum + chunk[ci] >= KSEL) {
        for (int b = ci * 64 + 63; b >= ci * 64; --b) {
          unsigned h = h1[b];
          if (cum + h >= KSEL) { B = b; above = cum; break; }
          cum += h;
        }
      } else cum += chunk[ci];
    }
    if (B < 0) { counters[3] = 0xFFFFFFFFu; counters[4] = 0u; }
    else       { counters[3] = (unsigned)B; counters[4] = above; }
  }
}

__global__ __launch_bounds__(256) void hist_lo(const unsigned* __restrict__ candKey,
                                               const unsigned* counters, unsigned* h2) {
  unsigned B = counters[3];
  if (B == 0xFFFFFFFFu) return;
  unsigned M = counters[0]; if (M > MCAP) M = MCAP;
  unsigned i = blockIdx.x * 256 + threadIdx.x;
  if (i < M) {
    unsigned k = candKey[i];
    if ((k >> 16) == B) atomicAdd(&h2[k & 0xFFFFu], 1u);
  }
}

__global__ __launch_bounds__(1024) void scan_lo(const unsigned* __restrict__ h2,
                                                unsigned* counters) {
  __shared__ unsigned chunk[1024];
  int t = threadIdx.x;
  unsigned s = 0;
  for (int i = 0; i < 64; ++i) s += h2[t * 64 + i];
  chunk[t] = s;
  __syncthreads();
  if (t == 0) {
    unsigned B = counters[3];
    if (B == 0xFFFFFFFFu) { counters[5] = 0u; counters[6] = 0u; return; }
    unsigned need = KSEL - counters[4];
    unsigned cum = 0; int L = -1; unsigned ties = 0;
    for (int ci = 1023; ci >= 0 && L < 0; --ci) {
      if (cum + chunk[ci] >= need) {
        for (int b = ci * 64 + 63; b >= ci * 64; --b) {
          unsigned h = h2[b];
          if (cum + h >= need) { L = b; ties = need - cum; break; }
          cum += h;
        }
      } else cum += chunk[ci];
    }
    counters[5] = (B << 16) | (unsigned)L;
    counters[6] = ties;
  }
}

__global__ __launch_bounds__(256) void select_k(const unsigned* __restrict__ candKey,
                                                const unsigned* __restrict__ candIdx,
                                                unsigned* counters, float* selScore,
                                                unsigned* selIdx) {
  unsigned M = counters[0]; if (M > MCAP) M = MCAP;
  unsigned T = counters[5], need = counters[6];
  unsigned i = blockIdx.x * 256 + threadIdx.x;
  if (i >= M) return;
  unsigned key = candKey[i];
  if (key > T) {
    unsigned p = atomicAdd(&counters[1], 1u);
    if (p < KSEL) { selScore[p] = __uint_as_float(key); selIdx[p] = candIdx[i]; }
  } else if (key == T && T != 0u) {
    unsigned tp = atomicAdd(&counters[2], 1u);
    if (tp < need) {
      unsigned p = atomicAdd(&counters[1], 1u);
      if (p < KSEL) { selScore[p] = __uint_as_float(key); selIdx[p] = candIdx[i]; }
    }
  }
}

// single-workgroup bitonic sort of 4096 (score desc, idx asc) in 32KB LDS
__global__ __launch_bounds__(1024) void sort_k(float* selScore, unsigned* selIdx) {
  __shared__ float    s[KSEL];
  __shared__ unsigned id[KSEL];
  int t = threadIdx.x;
  for (int e = t; e < KSEL; e += 1024) { s[e] = selScore[e]; id[e] = selIdx[e]; }
  __syncthreads();
  for (unsigned k = 2; k <= KSEL; k <<= 1) {
    for (unsigned j = k >> 1; j > 0; j >>= 1) {
      for (unsigned e = t; e < KSEL; e += 1024) {
        unsigned p = e ^ j;
        if (p > e) {
          float s1 = s[e], s2 = s[p];
          unsigned i1 = id[e], i2 = id[p];
          bool worse = (s1 < s2) || (s1 == s2 && i1 > i2);
          bool desc  = ((e & k) == 0);
          if (desc ? worse : !worse) { s[e] = s2; s[p] = s1; id[e] = i2; id[p] = i1; }
        }
      }
      __syncthreads();
    }
  }
  for (int e = t; e < KSEL; e += 1024) { selScore[e] = s[e]; selIdx[e] = id[e]; }
}

__global__ __launch_bounds__(256) void gather_boxes(const float* __restrict__ boxes,
                                                    const float* __restrict__ selScore,
                                                    const unsigned* __restrict__ selIdx,
                                                    float4* selBox, int* selLabel,
                                                    unsigned* keepArr) {
  int k = blockIdx.x * 256 + threadIdx.x;
  if (k >= KSEL) return;
  unsigned idx = selIdx[k];
  unsigned bi  = idx / NCLS;
  int      lab = (int)(idx % NCLS);
  const float* b = boxes + (size_t)bi * 4;
  float x1 = fminf(fmaxf(b[0], 0.0f), IMGW - 1.0f);
  float y1 = fminf(fmaxf(b[1], 0.0f), IMGH - 1.0f);
  float x2 = fminf(fmaxf(b[2], 0.0f), IMGW - 1.0f);
  float y2 = fminf(fmaxf(b[3], 0.0f), IMGH - 1.0f);
  selBox[k]   = make_float4(x1, y1, x2, y2);
  selLabel[k] = lab;
  keepArr[k]  = (selScore[k] > 0.0f) ? 1u : 0u;
}

// ---------------------------------------------------------------------------
// Fused tile kernel: WMMA one-hot Gram (same-class mask, K padded 81->96)
// + division-free IoU test, packed to a row-major bitmask via wave32 ballot.
// One wave per 16x(4*16) strip: A one-hot built once, reused for 4 column
// tiles.  C/D layout: lane L, acc[v] -> (row v+8*(L>>4), col L&15).
// ---------------------------------------------------------------------------
__global__ __launch_bounds__(32) void class_iou_mask(const float4* __restrict__ selBox,
                                                     const int* __restrict__ selLabel,
                                                     unsigned short* __restrict__ maskRows) {
  int ti  = blockIdx.x;        // 0..255 (row tile)
  int tj0 = blockIdx.y * 4;    // 0..252 (first of 4 column tiles)
  int lane = threadIdx.x;
  int rm = lane & 15;
  int hi = lane >> 4;

  __shared__ float4 rb[16];
  if (lane < 16) rb[lane] = selBox[ti * 16 + lane];
  int la = selLabel[ti * 16 + rm];   // A-matrix row label (M = lane&15)

  // Build A one-hot registers once per strip (16-bit A layout, K padded to 96)
  v16h a[3];
  #pragma unroll
  for (int chunk = 0; chunk < 3; ++chunk) {
    int kb = chunk * 32 + (hi ? 8 : 0);
    #pragma unroll
    for (int sIdx = 0; sIdx < 16; ++sIdx) {
      int v = sIdx >> 1, p = sIdx & 1;
      int K = kb + ((v & 4) ? 16 : 0) + ((v & 3) * 2) + p;
      a[chunk][sIdx] = (_Float16)((K == la) ? 1.0f : 0.0f);
    }
  }
  __syncthreads();

  for (int tj = tj0; tj < tj0 + 4; ++tj) {
    int lb = selLabel[tj * 16 + rm];       // B-matrix col label (N = lane&15)
    float4 bj = selBox[tj * 16 + rm];      // column box for this lane's N
    float areaB = (bj.z - bj.x) * (bj.w - bj.y);

    // S = onehot(labels_i) . onehot(labels_j)^T over K=96 (3 chunks of 32)
    v8f acc = {};
    #pragma unroll
    for (int chunk = 0; chunk < 3; ++chunk) {
      int kb = chunk * 32 + (hi ? 8 : 0);
      v16h b;
      #pragma unroll
      for (int sIdx = 0; sIdx < 16; ++sIdx) {
        int v = sIdx >> 1, p = sIdx & 1;
        int K = kb + ((v & 4) ? 16 : 0) + ((v & 3) * 2) + p;
        b[sIdx] = (_Float16)((K == lb) ? 1.0f : 0.0f);
      }
      acc = __builtin_amdgcn_wmma_f32_16x16x32_f16(false, a[chunk], false, b,
                                                   (short)0, acc, false, false);
    }

    #pragma unroll
    for (int v = 0; v < 8; ++v) {
      int r = v + (hi << 3);
      float4 bi = rb[r];
      float areaA = (bi.z - bi.x) * (bi.w - bi.y);
      float iw = fmaxf(fminf(bi.z, bj.z) - fmaxf(bi.x, bj.x), 0.0f);
      float ih = fmaxf(fminf(bi.w, bj.w) - fmaxf(bi.y, bj.y), 0.0f);
      float inter = iw * ih;
      // iou > T  <=>  inter > T*(areaA+areaB-inter+eps)   (union > 0)
      bool pred = (acc[v] > 0.5f) &&
                  (inter > NMS_T * (areaA + areaB - inter + 1e-9f));
      unsigned bal = (unsigned)__ballot(pred); // [15:0]=row v, [31:16]=row v+8
      if (lane == 0) {
        maskRows[(ti * 16 + v)     * 256 + tj] = (unsigned short)(bal & 0xFFFFu);
        maskRows[(ti * 16 + v + 8) * 256 + tj] = (unsigned short)(bal >> 16);
      }
    }
  }
}

// sequential greedy NMS: keep bitset (128 dwords) lives in LDS
__global__ __launch_bounds__(128) void nms_scan(const unsigned* __restrict__ mask,
                                                unsigned* keepArr) {
  __shared__ unsigned keepW[128];
  int t = threadIdx.x;
  unsigned w = 0;
  for (int b = 0; b < 32; ++b) w |= (keepArr[t * 32 + b] & 1u) << b;
  keepW[t] = w;
  __syncthreads();
  for (int i = 0; i < KSEL; ++i) {
    bool alive = (keepW[i >> 5] >> (i & 31)) & 1u;
    __syncthreads();
    if (alive) {
      unsigned m = mask[i * 128 + t];
      int wb = t * 32;
      unsigned gt;
      if (wb > i)            gt = 0xFFFFFFFFu;
      else if (wb + 31 <= i) gt = 0u;
      else                   gt = ~((2u << (i - wb)) - 1u);  // bits strictly > i
      keepW[t] &= ~(m & gt);
    }
    __syncthreads();
  }
  w = keepW[t];
  for (int b = 0; b < 32; ++b) keepArr[t * 32 + b] = (w >> b) & 1u;
}

__global__ __launch_bounds__(256) void write_out(const float4* __restrict__ selBox,
                                                 const float* __restrict__ selScore,
                                                 const int* __restrict__ selLabel,
                                                 const unsigned* __restrict__ keepArr,
                                                 float* __restrict__ out) {
  int k = blockIdx.x * 256 + threadIdx.x;
  if (k >= KSEL) return;
  float4 b = selBox[k];
  bool kp = keepArr[k] != 0u;
  out[k * 5 + 0] = kp ? b.x : 0.0f;
  out[k * 5 + 1] = kp ? b.y : 0.0f;
  out[k * 5 + 2] = kp ? b.z : 0.0f;
  out[k * 5 + 3] = kp ? b.w : 0.0f;
  out[k * 5 + 4] = kp ? selScore[k] : 0.0f;
  out[KSEL * 5 + k] = (float)selLabel[k];
  out[KSEL * 6 + k] = kp ? 1.0f : 0.0f;
}

// ---------------------------------------------------------------------------
extern "C" void kernel_launch(void* const* d_in, const int* in_sizes, int n_in,
                              void* d_out, int out_size, void* d_ws, size_t ws_size,
                              hipStream_t stream) {
  const float* x     = (const float*)d_in[0];
  const float* boxes = (const float*)d_in[1];
  float* out = (float*)d_out;
  char* ws = (char*)d_ws;

  // workspace layout (~7.4 MB total)
  unsigned*       counters = (unsigned*)(ws + 0);
  unsigned*       h1       = (unsigned*)(ws + 64);
  unsigned*       h2       = (unsigned*)(ws + 64 + 262144);          // 262208
  unsigned*       candKey  = (unsigned*)(ws + 524352);
  unsigned*       candIdx  = (unsigned*)(ws + 3014720);
  float*          selScore = (float*)   (ws + 5505088);
  unsigned*       selIdx   = (unsigned*)(ws + 5521472);
  float4*         selBox   = (float4*)  (ws + 5537856);
  int*            selLabel = (int*)     (ws + 5603392);
  unsigned*       keepArr  = (unsigned*)(ws + 5619776);
  unsigned short* maskU16  = (unsigned short*)(ws + 5636160);
  unsigned*       maskU32  = (unsigned*)(ws + 5636160);

  init_ws<<<256, 256, 0, stream>>>(counters, h1, h2, selScore, selIdx);
  softmax_compact<<<NROWS / 8, 256, 0, stream>>>(x, counters, candKey, candIdx);
  hist_hi<<<MCAP / 256, 256, 0, stream>>>(candKey, counters, h1);
  scan_hi<<<1, 1024, 0, stream>>>(h1, counters);
  hist_lo<<<MCAP / 256, 256, 0, stream>>>(candKey, counters, h2);
  scan_lo<<<1, 1024, 0, stream>>>(h2, counters);
  select_k<<<MCAP / 256, 256, 0, stream>>>(candKey, candIdx, counters, selScore, selIdx);
  sort_k<<<1, 1024, 0, stream>>>(selScore, selIdx);
  gather_boxes<<<KSEL / 256, 256, 0, stream>>>(boxes, selScore, selIdx,
                                               selBox, selLabel, keepArr);
  class_iou_mask<<<dim3(256, 64), 32, 0, stream>>>(selBox, selLabel, maskU16);
  nms_scan<<<1, 128, 0, stream>>>(maskU32, keepArr);
  write_out<<<KSEL / 256, 256, 0, stream>>>(selBox, selScore, selLabel, keepArr, out);
}
